// GraphMemoryNetwork_2_37280316129664
// MI455X (gfx1250) — matrix operand order repaired and checked by reading, not compiled, not run on hardware
//
#include <hip/hip_runtime.h>

#define N_NODES 50000
#define N_EDGES 640000
#define DIN     256
#define DD      128
#define NHEAD   8
#define DHD     16
#define NCLS    47
#define NLAYER  3
#define FF      512

typedef __attribute__((ext_vector_type(16))) __bf16 v16bf;
typedef __attribute__((ext_vector_type(8)))  float  v8f;

union V16U { unsigned short u[16]; uint4 q[2]; v16bf v; };

__device__ __forceinline__ unsigned short f2bf(float f) {
  unsigned int u = __float_as_uint(f);
  u += 0x7FFFu + ((u >> 16) & 1u);          // round-to-nearest-even
  return (unsigned short)(u >> 16);
}

__device__ __forceinline__ void atomicMaxF(float* addr, float val) {
  // int-max for positives, uint-min for negatives; valid with -inf init
  if (val >= 0.f) atomicMax((int*)addr, __float_as_int(val));
  else            atomicMin((unsigned int*)addr, __float_as_uint(val));
}

// ---------------------------------------------------------------------------
// f32 -> bf16 (row-major, used for the A-side activations)
// ---------------------------------------------------------------------------
__global__ __launch_bounds__(256)
void f32_to_bf16_kernel(const float* __restrict__ in,
                        unsigned short* __restrict__ out, long n) {
  long i = (long)blockIdx.x * blockDim.x + threadIdx.x;
  if (i < n) out[i] = f2bf(in[i]);
}

// ---------------------------------------------------------------------------
// Pack weight W[K,Nn] (f32, row-major) into WMMA B-fragment order (bf16):
//   out[((nt*ksteps + ks)*32 + lane)*16 + e]
//   col = nt*16 + (lane&15), k = ks*32 + (lane>>4)*16 + e
// Columns >= Nn are zero-padded (padN = Nn rounded up to 16).
// ---------------------------------------------------------------------------
__global__ __launch_bounds__(256)
void pack_b_kernel(const float* __restrict__ W, unsigned short* __restrict__ out,
                   int K, int Nn, int padN) {
  const long total = (long)padN * K;
  long i = (long)blockIdx.x * blockDim.x + threadIdx.x;
  if (i >= total) return;
  const int ksteps = K >> 5;
  const int e    = (int)(i & 15);
  const int lane = (int)((i >> 4) & 31);
  const int ks   = (int)((i >> 9) % ksteps);
  const int nt   = (int)(i / ((long)ksteps << 9));
  const int col  = nt * 16 + (lane & 15);
  const int k    = ks * 32 + (lane >> 4) * 16 + e;
  out[i] = (col < Nn) ? f2bf(W[(long)k * Nn + col]) : (unsigned short)0;
}

// ---------------------------------------------------------------------------
// Wave-tile WMMA GEMM: C[M,Nn] = A[M,K](bf16 rowmajor) @ Bp(packed bf16)
// block = 256 threads = 8 waves; wave owns one 16-row M-tile and NT
// consecutive 16-col N-tiles (independent accumulators -> XDL ILP).
// Inner loop: 2 coalesced b128 A-loads + 2*NT coalesced b128 B-loads + NT WMMA.
// ---------------------------------------------------------------------------
template <int NT>
__global__ __launch_bounds__(256)
void gemm_bf16_kernel(const unsigned short* __restrict__ A,
                      const unsigned short* __restrict__ Bp,
                      const float* __restrict__ bias,
                      float* __restrict__ C,
                      unsigned short* __restrict__ Cbf,
                      int M, int Nn, int K, int relu) {
  const int lane  = threadIdx.x & 31;
  const int wv    = threadIdx.x >> 5;
  const int mtile = blockIdx.y * 8 + wv;
  if (mtile * 16 >= M) return;

  const int hi     = lane >> 4;                   // lane half (K sub-range)
  const int ksteps = K >> 5;

  const unsigned short* arow  = A + (long)(mtile * 16 + (lane & 15)) * K;
  const unsigned short* bbase = Bp + ((long)(blockIdx.x * NT) * ksteps * 512) + lane * 16;

  v8f acc[NT];
#pragma unroll
  for (int t = 0; t < NT; ++t) acc[t] = (v8f){};

  for (int ks = 0; ks < ksteps; ++ks) {
    V16U a;
    a.q[0] = *(const uint4*)(arow + ks * 32 + hi * 16);
    a.q[1] = *(const uint4*)(arow + ks * 32 + hi * 16 + 8);
#pragma unroll
    for (int t = 0; t < NT; ++t) {
      const unsigned short* bp = bbase + ((long)t * ksteps + ks) * 512;
      V16U b;
      b.q[0] = *(const uint4*)(bp);
      b.q[1] = *(const uint4*)(bp + 8);
      acc[t] = __builtin_amdgcn_wmma_f32_16x16x32_bf16(
          false, a.v, false, b.v, (short)0, acc[t], false, false);
    }
  }

#pragma unroll
  for (int t = 0; t < NT; ++t) {
    const int ccol = (blockIdx.x * NT + t) * 16 + (lane & 15);
    if (ccol < Nn) {
      const float bv = bias ? bias[ccol] : 0.f;
#pragma unroll
      for (int r = 0; r < 8; ++r) {
        const int rr = mtile * 16 + hi * 8 + r;   // C row per VGPR layout
        float v = acc[t][r] + bv;
        if (relu) v = fmaxf(v, 0.f);
        C[(long)rr * Nn + ccol] = v;
        if (Cbf) Cbf[(long)rr * Nn + ccol] = f2bf(v);
      }
    }
  }
}

// ---------------------------------------------------------------------------
// LayerNorm over D=128 per row; one wave32 per row, 4 floats per lane.
// Optional residual input; writes f32 + bf16 outputs.
// ---------------------------------------------------------------------------
__global__ __launch_bounds__(256)
void layernorm_kernel(const float* __restrict__ x, const float* __restrict__ res,
                      const float* __restrict__ g, const float* __restrict__ b,
                      float* __restrict__ hout, unsigned short* __restrict__ hbf,
                      int n) {
  const int rowi = (int)((blockIdx.x * blockDim.x + threadIdx.x) >> 5);
  const int lane = threadIdx.x & 31;
  if (rowi >= n) return;

  float4 val = ((const float4*)(x + (long)rowi * DD))[lane];
  if (res) {
    float4 r = ((const float4*)(res + (long)rowi * DD))[lane];
    val.x += r.x; val.y += r.y; val.z += r.z; val.w += r.w;
  }
  float s  = val.x + val.y + val.z + val.w;
  float ss = val.x*val.x + val.y*val.y + val.z*val.z + val.w*val.w;
#pragma unroll
  for (int off = 16; off > 0; off >>= 1) {
    s  += __shfl_xor(s,  off, 32);
    ss += __shfl_xor(ss, off, 32);
  }
  const float mean = s * (1.f / DD);
  const float var  = ss * (1.f / DD) - mean * mean;
  const float rstd = rsqrtf(var + 1e-5f);

  const int col = lane * 4;
  float4 gg = *(const float4*)(g + col);
  float4 bb = *(const float4*)(b + col);
  float4 y;
  y.x = (val.x - mean) * rstd * gg.x + bb.x;
  y.y = (val.y - mean) * rstd * gg.y + bb.y;
  y.z = (val.z - mean) * rstd * gg.z + bb.z;
  y.w = (val.w - mean) * rstd * gg.w + bb.w;
  ((float4*)(hout + (long)rowi * DD))[lane] = y;
  if (hbf) {
    ushort4 o;
    o.x = f2bf(y.x); o.y = f2bf(y.y); o.z = f2bf(y.z); o.w = f2bf(y.w);
    ((ushort4*)(hbf + (long)rowi * DD))[lane] = o;
  }
}

// ---------------------------------------------------------------------------
// Per-layer attention scratch init: nmax=-inf, denom=0, agg=0
// ---------------------------------------------------------------------------
__global__ __launch_bounds__(256)
void init_attn_kernel(float* __restrict__ nmax, float* __restrict__ denom,
                      float* __restrict__ agg) {
  long i = (long)blockIdx.x * blockDim.x + threadIdx.x;
  if (i < (long)N_NODES * NHEAD) { nmax[i] = -INFINITY; denom[i] = 0.f; }
  if (i < (long)N_NODES * DD)    { agg[i] = 0.f; }
}

// ---------------------------------------------------------------------------
// Pass 1: per-(edge,head) score = <q[dst,h], k[src,h]> * scale; atomic-max.
// ---------------------------------------------------------------------------
__global__ __launch_bounds__(256)
void edge_scores_kernel(const float* __restrict__ q, const float* __restrict__ k,
                        const int* __restrict__ src, const int* __restrict__ dst,
                        float scale, float* __restrict__ sbuf,
                        float* __restrict__ nmax) {
  long i = (long)blockIdx.x * blockDim.x + threadIdx.x;
  if (i >= (long)N_EDGES * NHEAD) return;
  const int e  = (int)(i >> 3);          // / NHEAD
  const int hh = (int)(i & (NHEAD - 1));
  const int s  = src[e];
  const int d  = dst[e];
  const float4* qp = (const float4*)(q + (long)d * DD + hh * DHD);
  const float4* kp = (const float4*)(k + (long)s * DD + hh * DHD);
  float acc = 0.f;
#pragma unroll
  for (int j = 0; j < 4; ++j) {
    float4 a = qp[j], b = kp[j];
    acc += a.x*b.x + a.y*b.y + a.z*b.z + a.w*b.w;
  }
  acc *= scale;
  sbuf[i] = acc;
  atomicMaxF(&nmax[(long)d * NHEAD + hh], acc);
}

// ---------------------------------------------------------------------------
// Pass 2: a = exp(s - max); atomic denom sum.
// ---------------------------------------------------------------------------
__global__ __launch_bounds__(256)
void edge_exp_kernel(float* __restrict__ sbuf, const float* __restrict__ nmax,
                     float* __restrict__ denom, const int* __restrict__ dst) {
  long i = (long)blockIdx.x * blockDim.x + threadIdx.x;
  if (i >= (long)N_EDGES * NHEAD) return;
  const int e  = (int)(i >> 3);
  const int hh = (int)(i & (NHEAD - 1));
  const int d  = dst[e];
  const float a = __expf(sbuf[i] - nmax[(long)d * NHEAD + hh]);
  sbuf[i] = a;
  atomicAdd(&denom[(long)d * NHEAD + hh], a);
}

// ---------------------------------------------------------------------------
// Pass 3: agg[dst] += (a/denom) * v[src]
// ---------------------------------------------------------------------------
__global__ __launch_bounds__(256)
void edge_agg_kernel(const float* __restrict__ sbuf, const float* __restrict__ denom,
                     const float* __restrict__ v, const int* __restrict__ src,
                     const int* __restrict__ dst, float* __restrict__ agg) {
  long i = (long)blockIdx.x * blockDim.x + threadIdx.x;
  if (i >= (long)N_EDGES * NHEAD) return;
  const int e  = (int)(i >> 3);
  const int hh = (int)(i & (NHEAD - 1));
  const int s  = src[e];
  const int d  = dst[e];
  const float w = sbuf[i] / denom[(long)d * NHEAD + hh];
  const float4* vp = (const float4*)(v + (long)s * DD + hh * DHD);
  float* ap = agg + (long)d * DD + hh * DHD;
#pragma unroll
  for (int j = 0; j < 4; ++j) {
    float4 t = vp[j];
    atomicAdd(ap + j*4 + 0, w * t.x);
    atomicAdd(ap + j*4 + 1, w * t.y);
    atomicAdd(ap + j*4 + 2, w * t.z);
    atomicAdd(ap + j*4 + 3, w * t.w);
  }
}

// ---------------------------------------------------------------------------
// Host orchestration
// ---------------------------------------------------------------------------
static inline long cdivl(long a, long b) { return (a + b - 1) / b; }

extern "C" void kernel_launch(void* const* d_in, const int* in_sizes, int n_in,
                              void* d_out, int out_size, void* d_ws, size_t ws_size,
                              hipStream_t stream) {
  const float* X        = (const float*)d_in[0];
  const int*   eidx     = (const int*)d_in[1];
  const float* emb_w    = (const float*)d_in[2];
  const float* emb_b    = (const float*)d_in[3];
  const float* emb_ln_g = (const float*)d_in[4];
  const float* emb_ln_b = (const float*)d_in[5];
  const float* Wq       = (const float*)d_in[6];
  const float* Wk       = (const float*)d_in[7];
  const float* Wv       = (const float*)d_in[8];
  const float* Wo       = (const float*)d_in[9];
  const float* ln_g     = (const float*)d_in[10];
  const float* ln_b     = (const float*)d_in[11];
  const float* f1_w     = (const float*)d_in[12];
  const float* f1_b     = (const float*)d_in[13];
  const float* f2_w     = (const float*)d_in[14];
  const float* f2_b     = (const float*)d_in[15];
  const float* f3_w     = (const float*)d_in[16];
  const float* f3_b     = (const float*)d_in[17];
  float* out = (float*)d_out;

  const int* src = eidx;            // edge_index[0]
  const int* dst = eidx + N_EDGES;  // edge_index[1]

  const int NC_PAD = 48;            // NCLS padded to tile multiple

  // ---- carve workspace ----
  char* p = (char*)d_ws;
  auto alloc = [&](size_t bytes) -> char* {
    char* r = p; p += (bytes + 255) & ~(size_t)255; return r;
  };
  unsigned short* Xbf    = (unsigned short*)alloc((size_t)N_NODES * DIN * 2);
  unsigned short* embwP  = (unsigned short*)alloc((size_t)DIN * DD * 2);
  unsigned short* WqP    = (unsigned short*)alloc((size_t)NLAYER * DD * DD * 2);
  unsigned short* WkP    = (unsigned short*)alloc((size_t)NLAYER * DD * DD * 2);
  unsigned short* WvP    = (unsigned short*)alloc((size_t)NLAYER * DD * DD * 2);
  unsigned short* WoP    = (unsigned short*)alloc((size_t)NLAYER * DD * DD * 2);
  unsigned short* f1P    = (unsigned short*)alloc((size_t)DD * FF * 2);
  unsigned short* f2P    = (unsigned short*)alloc((size_t)FF * FF * 2);
  unsigned short* f3P    = (unsigned short*)alloc((size_t)FF * NC_PAD * 2);
  float*          h      = (float*)alloc((size_t)N_NODES * DD * 4);
  unsigned short* hbf    = (unsigned short*)alloc((size_t)N_NODES * DD * 2);
  float*          tmp    = (float*)alloc((size_t)N_NODES * DD * 4);
  float*          qb     = (float*)alloc((size_t)N_NODES * DD * 4);
  float*          kb     = (float*)alloc((size_t)N_NODES * DD * 4);
  float*          vb     = (float*)alloc((size_t)N_NODES * DD * 4);
  float*          sbuf   = (float*)alloc((size_t)N_EDGES * NHEAD * 4);
  float*          nmax   = (float*)alloc((size_t)N_NODES * NHEAD * 4);
  float*          denom  = (float*)alloc((size_t)N_NODES * NHEAD * 4);
  float*          agg    = (float*)alloc((size_t)N_NODES * DD * 4);
  unsigned short* aggbf  = (unsigned short*)alloc((size_t)N_NODES * DD * 2);
  float*          t1     = (float*)alloc((size_t)N_NODES * FF * 4);
  unsigned short* t1bf   = (unsigned short*)alloc((size_t)N_NODES * FF * 2);
  float*          t2     = (float*)alloc((size_t)N_NODES * FF * 4);
  unsigned short* t2bf   = (unsigned short*)alloc((size_t)N_NODES * FF * 2);

  auto cvt = [&](const float* in, unsigned short* o, long n) {
    f32_to_bf16_kernel<<<dim3((unsigned)cdivl(n, 256)), 256, 0, stream>>>(in, o, n);
  };
  auto packB = [&](const float* W, unsigned short* o, int K, int Nn, int padN) {
    long total = (long)padN * K;
    pack_b_kernel<<<dim3((unsigned)cdivl(total, 256)), 256, 0, stream>>>(W, o, K, Nn, padN);
  };
  auto gemm4 = [&](const unsigned short* A, const unsigned short* Bp,
                   const float* bias, float* C, unsigned short* Cbf,
                   int M, int Nn, int padN, int K, int relu) {
    dim3 grid((unsigned)(padN / 64), (unsigned)cdivl(cdivl(M, 16), 8));
    gemm_bf16_kernel<4><<<grid, 256, 0, stream>>>(A, Bp, bias, C, Cbf, M, Nn, K, relu);
  };
  auto gemm3 = [&](const unsigned short* A, const unsigned short* Bp,
                   const float* bias, float* C, unsigned short* Cbf,
                   int M, int Nn, int padN, int K, int relu) {
    dim3 grid((unsigned)(padN / 48), (unsigned)cdivl(cdivl(M, 16), 8));
    gemm_bf16_kernel<3><<<grid, 256, 0, stream>>>(A, Bp, bias, C, Cbf, M, Nn, K, relu);
  };
  auto lnorm = [&](const float* x, const float* res, const float* g,
                   const float* b, float* ho, unsigned short* hb) {
    layernorm_kernel<<<dim3((unsigned)cdivl((long)N_NODES * 32, 256)), 256, 0, stream>>>(
        x, res, g, b, ho, hb, N_NODES);
  };

  // ---- convert A-side input + pack all weights into fragment layout ----
  cvt(X, Xbf, (long)N_NODES * DIN);
  packB(emb_w, embwP, DIN, DD, DD);
  for (int l = 0; l < NLAYER; ++l) {
    const size_t wo = (size_t)l * DD * DD;
    packB(Wq + wo, WqP + wo, DD, DD, DD);
    packB(Wk + wo, WkP + wo, DD, DD, DD);
    packB(Wv + wo, WvP + wo, DD, DD, DD);
    packB(Wo + wo, WoP + wo, DD, DD, DD);
  }
  packB(f1_w, f1P, DD, FF, FF);
  packB(f2_w, f2P, FF, FF, FF);
  packB(f3_w, f3P, FF, NCLS, NC_PAD);

  // ---- embedding: relu(X @ emb_w + b) -> LayerNorm ----
  gemm4(Xbf, embwP, emb_b, tmp, nullptr, N_NODES, DD, DD, DIN, /*relu=*/1);
  lnorm(tmp, nullptr, emb_ln_g, emb_ln_b, h, hbf);

  const float scale = 1.f / sqrtf((float)DHD);
  const long EH = (long)N_EDGES * NHEAD;
  const unsigned edge_blocks = (unsigned)cdivl(EH, 256);
  const unsigned init_blocks = (unsigned)cdivl((long)N_NODES * DD, 256);

  for (int l = 0; l < NLAYER; ++l) {
    const size_t wo = (size_t)l * DD * DD;
    gemm4(hbf, WqP + wo, nullptr, qb, nullptr, N_NODES, DD, DD, DD, 0);
    gemm4(hbf, WkP + wo, nullptr, kb, nullptr, N_NODES, DD, DD, DD, 0);
    gemm4(hbf, WvP + wo, nullptr, vb, nullptr, N_NODES, DD, DD, DD, 0);

    init_attn_kernel<<<init_blocks, 256, 0, stream>>>(nmax, denom, agg);
    edge_scores_kernel<<<edge_blocks, 256, 0, stream>>>(qb, kb, src, dst, scale, sbuf, nmax);
    edge_exp_kernel<<<edge_blocks, 256, 0, stream>>>(sbuf, nmax, denom, dst);
    edge_agg_kernel<<<edge_blocks, 256, 0, stream>>>(sbuf, denom, vb, src, dst, agg);

    cvt(agg, aggbf, (long)N_NODES * DD);
    gemm4(aggbf, WoP + wo, nullptr, tmp, nullptr, N_NODES, DD, DD, DD, 0);
    lnorm(tmp, /*res=*/h, ln_g + (size_t)l * DD, ln_b + (size_t)l * DD, h, hbf);
  }

  // ---- FFN head ----
  gemm4(hbf,  f1P, f1_b, t1,  t1bf, N_NODES, FF, FF, DD, 1);
  gemm4(t1bf, f2P, f2_b, t2,  t2bf, N_NODES, FF, FF, FF, 1);
  gemm3(t2bf, f3P, f3_b, out, nullptr, N_NODES, NCLS, NC_PAD, FF, 0);
}
